// Router_34497177321800
// MI455X (gfx1250) — compile-verified, matrix operand order checked by reading, not run on hardware
//
#include <hip/hip_runtime.h>
#include <hip/hip_bf16.h>
#include <math.h>
#include <stdint.h>

#define NUM_EXPERTS 8
#define D_MODEL 512
#define D_FF 2048
#define T_TOKENS 16384   /* 8 * 2048 */
#define BM 64            /* token rows per block tile  */
#define BFF 64           /* FF chunk                    */
#define NCHUNK (D_FF / BFF)      /* 32  */
#define MAXTILE (T_TOKENS / BM)  /* 256 */
#define WBLK_BYTES (BFF * D_MODEL * 2)   /* 64KB per packed weight chunk */

typedef __attribute__((ext_vector_type(16))) __bf16 v16bf;
typedef __attribute__((ext_vector_type(8)))  __bf16 v8bf;
typedef __attribute__((ext_vector_type(8)))  float  v8f;
typedef __attribute__((ext_vector_type(4)))  unsigned v4u;
typedef __attribute__((ext_vector_type(8)))  unsigned v8u;

// ---------------------------------------------------------------------------
// Tensor Data Mover: one descriptor moves a contiguous 64KB block
// (1 row, tile_dim0 = 32768 x 2B elements) from global to LDS.
// Group0: count=1 | lds_addr | global_addr | type=2.  Group1: data_size=2B,
// tensor_dim0=tile_dim0=32768, tensor_dim1=tile_dim1=1, stride0=32768.
// Issue from a single wave; tracked by TENSORcnt.
// ---------------------------------------------------------------------------
__device__ __forceinline__ void tdm_load_64kb(unsigned lds_off, const void* gptr)
{
    uint64_t ga = (uint64_t)gptr;
    v4u g0;
    g0[0] = 1u;                                     // count=1, user desc
    g0[1] = lds_off;                                // lds_addr (bytes)
    g0[2] = (unsigned)ga;                           // global_addr[31:0]
    g0[3] = (unsigned)(ga >> 32) | (2u << 30);      // global_addr[56:32] | type=2
    v8u g1;
    g1[0] = 1u << 16;        // workgroup_mask=0, data_size=1 (2 bytes)
    g1[1] = 0x80000000u;     // tensor_dim0[15:0]=32768 in bits 63:48
    g1[2] = 0x00010000u;     // tensor_dim0[31:16]=0, tensor_dim1[15:0]=1
    g1[3] = 0x80000000u;     // tensor_dim1[31:16]=0, tile_dim0=32768
    g1[4] = 0x00000001u;     // tile_dim1=1, tile_dim2=0
    g1[5] = 0x00008000u;     // tensor_dim0_stride[31:0]=32768
    g1[6] = 0u;              // stride0 hi, stride1 lo
    g1[7] = 0u;              // stride1 hi
    asm volatile("tensor_load_to_lds %0, %1" :: "s"(g0), "s"(g1) : "memory");
}

__device__ __forceinline__ void wait_tensor0()
{
#if __has_builtin(__builtin_amdgcn_s_wait_tensorcnt)
    __builtin_amdgcn_s_wait_tensorcnt(0);
#else
    asm volatile("s_wait_tensorcnt 0x0" ::: "memory");
#endif
}

__device__ __forceinline__ unsigned lds_off_of(const void* p)
{
    return (unsigned)(size_t)p;   // flat LDS addr = {SHARED_BASE, offset}
}

// ---------------------------------------------------------------------------
// Stage 0 (packed path): one-shot fp32 -> bf16 weight repack into LDS-tile
// order so each FFN weight chunk is one contiguous 64KB TDM transfer.
//   pw1[((e*32+c)*64 + n)*512 + k]  = w1[e][k][c*64+n]
//   pw2[((e*32+c)*512 + n)*64 + k2] = w2[e][c*64+k2][n]
// ---------------------------------------------------------------------------
__global__ __launch_bounds__(256) void pack_w1_kernel(
    const float* __restrict__ w1, __bf16* __restrict__ pw1)
{
    size_t l = ((size_t)blockIdx.x * 256 + threadIdx.x) * 4;   // [e][k][ff]
    int ff = (int)(l & (D_FF - 1));
    size_t ek = l >> 11;
    int k = (int)(ek & (D_MODEL - 1));
    int e = (int)(ek >> 9);
    float4 wv = *(const float4*)(w1 + l);
    int c = ff >> 6, n0 = ff & 63;
    __bf16* dst = pw1 + ((size_t)(e * NCHUNK + c) * BFF + n0) * D_MODEL + k;
    dst[0 * D_MODEL] = (__bf16)wv.x;
    dst[1 * D_MODEL] = (__bf16)wv.y;
    dst[2 * D_MODEL] = (__bf16)wv.z;
    dst[3 * D_MODEL] = (__bf16)wv.w;
}

__global__ __launch_bounds__(256) void pack_w2_kernel(
    const float* __restrict__ w2, __bf16* __restrict__ pw2)
{
    size_t l = ((size_t)blockIdx.x * 256 + threadIdx.x) * 4;   // [e][ff][d]
    int d = (int)(l & (D_MODEL - 1));
    size_t eff = l >> 9;
    int ff = (int)(eff & (D_FF - 1));
    int e = (int)(eff >> 11);
    float4 wv = *(const float4*)(w2 + l);
    int c = ff >> 6, k2 = ff & 63;
    __bf16* dst = pw2 + ((size_t)(e * NCHUNK + c) * D_MODEL + d) * BFF + k2;
    dst[0 * BFF] = (__bf16)wv.x;
    dst[1 * BFF] = (__bf16)wv.y;
    dst[2 * BFF] = (__bf16)wv.z;
    dst[3 * BFF] = (__bf16)wv.w;
}

// ---------------------------------------------------------------------------
// Stage 1: gate logits, top-2 softmax, compaction into per-expert token lists
// ---------------------------------------------------------------------------
__global__ __launch_bounds__(256) void moe_gate_kernel(
    const float* __restrict__ x, const float* __restrict__ gate_w,
    int* __restrict__ cnt, int* __restrict__ idxl, float* __restrict__ wgtl)
{
    int t = blockIdx.x * blockDim.x + threadIdx.x;
    if (t >= T_TOKENS) return;
    const float* xr = x + (size_t)t * D_MODEL;

    float acc[NUM_EXPERTS];
#pragma unroll
    for (int e = 0; e < NUM_EXPERTS; ++e) acc[e] = 0.f;

    for (int d = 0; d < D_MODEL; d += 4) {
        float4 xv = *(const float4*)(xr + d);
#pragma unroll
        for (int e = 0; e < NUM_EXPERTS; ++e) {
            float4 gv = *(const float4*)(gate_w + e * D_MODEL + d);
            acc[e] += xv.x * gv.x + xv.y * gv.y + xv.z * gv.z + xv.w * gv.w;
        }
    }
    int e1 = 0; float v1 = acc[0];
#pragma unroll
    for (int e = 1; e < NUM_EXPERTS; ++e)
        if (acc[e] > v1) { v1 = acc[e]; e1 = e; }
    int e2 = (e1 == 0) ? 1 : 0; float v2 = acc[e2];
#pragma unroll
    for (int e = 0; e < NUM_EXPERTS; ++e)
        if (e != e1 && acc[e] > v2) { v2 = acc[e]; e2 = e; }

    float w1s = 1.f / (1.f + __expf(v2 - v1));
    float w2s = 1.f - w1s;

    int s1 = atomicAdd(&cnt[e1], 1);
    idxl[e1 * T_TOKENS + s1] = t;  wgtl[e1 * T_TOKENS + s1] = w1s;
    int s2 = atomicAdd(&cnt[e2], 1);
    idxl[e2 * T_TOKENS + s2] = t;  wgtl[e2 * T_TOKENS + s2] = w2s;
}

// ---------------------------------------------------------------------------
// WMMA fragment: two contiguous 16B LDS reads per CDNA5 16-bit A/B layout
// (lanes 0-15: K 0-7 & 16-23; lanes 16-31: K 8-15 & 24-31).
// ---------------------------------------------------------------------------
__device__ __forceinline__ v16bf load_frag(const __bf16* row_ptr, int kk, int kh)
{
    v8bf lo = *(const v8bf*)(row_ptr + kk + 8 * kh);
    v8bf hi = *(const v8bf*)(row_ptr + kk + 16 + 8 * kh);
    v16bf f;
#pragma unroll
    for (int q = 0; q < 8; ++q) { f[q] = lo[q]; f[q + 8] = hi[q]; }
    return f;
}

// fallback staging (fp32 global -> bf16 LDS, VALU convert)
__device__ __forceinline__ void stage_w1_cvt(const float* __restrict__ w1e,
                                             __bf16* s_dst, int ff0, int tid)
{
#pragma unroll
    for (int i = 0; i < 32; ++i) {
        int l = i * 1024 + tid * 4;
        int k = l >> 6, n = l & 63;
        float4 wv = *(const float4*)(w1e + (size_t)k * D_FF + ff0 + n);
        s_dst[(n + 0) * D_MODEL + k] = (__bf16)wv.x;
        s_dst[(n + 1) * D_MODEL + k] = (__bf16)wv.y;
        s_dst[(n + 2) * D_MODEL + k] = (__bf16)wv.z;
        s_dst[(n + 3) * D_MODEL + k] = (__bf16)wv.w;
    }
}

__device__ __forceinline__ void stage_w2_cvt(const float* __restrict__ w2e,
                                             __bf16* s_dst, int ff0, int tid)
{
#pragma unroll
    for (int i = 0; i < 32; ++i) {
        int l = i * 1024 + tid * 4;
        int k2 = l >> 9, n = l & 511;
        float4 wv = *(const float4*)(w2e + (size_t)(ff0 + k2) * D_MODEL + n);
        s_dst[(n + 0) * BFF + k2] = (__bf16)wv.x;
        s_dst[(n + 1) * BFF + k2] = (__bf16)wv.y;
        s_dst[(n + 2) * BFF + k2] = (__bf16)wv.z;
        s_dst[(n + 3) * BFF + k2] = (__bf16)wv.w;
    }
}

// ---------------------------------------------------------------------------
// Stage 2: fused expert FFN.  Block = (expert e, 64-token tile of its list).
// PACKED: weight chunks staged by TDM (tensor_load_to_lds), double-buffered
// so transfers overlap WMMA compute:
//   prologue: TDM W1(0)->A
//   loop c:   TDM W2(c)->B | GEMM1(A) | wait+bar |
//             TDM W1(c+1)->A | GEMM2(B) | wait+bar
// ---------------------------------------------------------------------------
template<bool PACKED>
__global__ __launch_bounds__(256) void moe_ffn_kernel(
    const float* __restrict__ x,
    const float* __restrict__ w1, const float* __restrict__ b1,
    const float* __restrict__ w2, const float* __restrict__ b2,
    const __bf16* __restrict__ pw1, const __bf16* __restrict__ pw2,
    const int* __restrict__ cnt, const int* __restrict__ idxl,
    const float* __restrict__ wgtl, float* __restrict__ out)
{
    __shared__ __bf16 s_x [BM * D_MODEL];    // [m][k]  64x512  (64 KB)
    __shared__ __bf16 s_wA[BFF * D_MODEL];   // W1T [n][k]      (64 KB)
    __shared__ __bf16 s_wB[BFF * D_MODEL];   // W2T [n][k2]     (64 KB)
    __shared__ __bf16 s_h [BM * BFF];        // [m][ff] 64x64   (8 KB)
    __shared__ int    s_tok[BM];
    __shared__ float  s_wgt[BM];

    const int e    = blockIdx.y;
    const int tile = blockIdx.x;
    const int n_e  = cnt[e];
    if (tile * BM >= n_e) return;

    const int tid  = threadIdx.x;
    const int wave = tid >> 5;
    const int lane = tid & 31;
    const int l15  = lane & 15;
    const int kh   = lane >> 4;
    const unsigned offA = lds_off_of(s_wA);
    const unsigned offB = lds_off_of(s_wB);

    const float*  w1e  = w1 + (size_t)e * D_MODEL * D_FF;
    const float*  w2e  = w2 + (size_t)e * D_FF * D_MODEL;
    const char*   pw1e = (const char*)pw1 + (size_t)e * NCHUNK * WBLK_BYTES;
    const char*   pw2e = (const char*)pw2 + (size_t)e * NCHUNK * WBLK_BYTES;

    if (tid < BM) {
        int slot = tile * BM + tid;
        int tok = 0; float wg = 0.f;
        if (slot < n_e) { tok = idxl[e * T_TOKENS + slot]; wg = wgtl[e * T_TOKENS + slot]; }
        s_tok[tid] = tok; s_wgt[tid] = wg;
    }

    // prologue prefetch: W1(0) -> A (overlaps the X gather below)
    if (PACKED && wave == 0) tdm_load_64kb(offA, pw1e);
    __syncthreads();                     // s_tok visible

    // gather + convert X tile (64x512 fp32 -> bf16), once per block
#pragma unroll
    for (int i = 0; i < 32; ++i) {
        int l = i * 1024 + tid * 4;
        int m = l >> 9, k = l & 511;
        float4 xv = *(const float4*)(x + (size_t)s_tok[m] * D_MODEL + k);
        __bf16* dst = &s_x[m * D_MODEL + k];
        dst[0] = (__bf16)xv.x; dst[1] = (__bf16)xv.y;
        dst[2] = (__bf16)xv.z; dst[3] = (__bf16)xv.w;
    }
    if (!PACKED) stage_w1_cvt(w1e, s_wA, 0, tid);
    if (PACKED && wave == 0) wait_tensor0();
    __syncthreads();                     // X + W1(0) ready

    v8f accY[16];
#pragma unroll
    for (int i = 0; i < 16; ++i) accY[i] = {};

    for (int c = 0; c < NCHUNK; ++c) {
        const int ff0 = c * BFF;

        // 1. prefetch W2(c) -> B (overlaps GEMM1)
        if (PACKED) { if (wave == 0) tdm_load_64kb(offB, pw2e + (size_t)c * WBLK_BYTES); }
        else        stage_w2_cvt(w2e, s_wB, ff0, tid);

        // 2. GEMM1: X(64x512) @ W1c(512x64) -> H, 2 tiles/wave, fused bias+ReLU
#pragma unroll
        for (int j = 0; j < 2; ++j) {
            int tt = wave * 2 + j;
            int mt = tt >> 2, nt = tt & 3;
            v8f acc = {};
            const __bf16* arow = &s_x [(mt * 16 + l15) * D_MODEL];
            const __bf16* brow = &s_wA[(nt * 16 + l15) * D_MODEL];
#pragma unroll
            for (int kk = 0; kk < D_MODEL; kk += 32) {
                v16bf a = load_frag(arow, kk, kh);
                v16bf b = load_frag(brow, kk, kh);
                acc = __builtin_amdgcn_wmma_f32_16x16x32_bf16(
                        false, a, false, b, (short)0, acc, false, false);
            }
            int ncol = nt * 16 + l15;
            float bb = b1[e * D_FF + ff0 + ncol];
#pragma unroll
            for (int r = 0; r < 8; ++r) {
                int m = mt * 16 + kh * 8 + r;
                float hv = acc[r] + bb;
                hv = hv > 0.f ? hv : 0.f;
                s_h[m * BFF + ncol] = (__bf16)hv;
            }
        }

        // 3. W2 + H visible to everyone; all waves done reading A
        if (PACKED && wave == 0) wait_tensor0();
        __syncthreads();

        // 4. prefetch W1(c+1) -> A (overlaps GEMM2)
        if (c + 1 < NCHUNK) {
            if (PACKED) { if (wave == 0) tdm_load_64kb(offA, pw1e + (size_t)(c + 1) * WBLK_BYTES); }
            else        stage_w1_cvt(w1e, s_wA, ff0 + BFF, tid);
        }

        // 5. GEMM2: H(64x64) @ W2c(64x512) -> Y accum, 16 tiles/wave
#pragma unroll
        for (int mt = 0; mt < 4; ++mt) {
            const __bf16* arow = &s_h[(mt * 16 + l15) * BFF];
#pragma unroll
            for (int nq = 0; nq < 4; ++nq) {
                int nt = wave * 4 + nq;
                const __bf16* brow = &s_wB[(nt * 16 + l15) * BFF];
                v8f acc = accY[mt * 4 + nq];
#pragma unroll
                for (int kk = 0; kk < BFF; kk += 32) {
                    v16bf a = load_frag(arow, kk, kh);
                    v16bf b = load_frag(brow, kk, kh);
                    acc = __builtin_amdgcn_wmma_f32_16x16x32_bf16(
                            false, a, false, b, (short)0, acc, false, false);
                }
                accY[mt * 4 + nq] = acc;
            }
        }

        // 6. W1(c+1) ready; B and H free for next iteration
        if (PACKED && wave == 0) wait_tensor0();
        __syncthreads();
    }

    // epilogue: out[tok] += wgt * (Y + b2)  (tokens live in 2 expert lists)
#pragma unroll
    for (int mt = 0; mt < 4; ++mt) {
#pragma unroll
        for (int nq = 0; nq < 4; ++nq) {
            int nt = wave * 4 + nq;
            int ncol = nt * 16 + l15;
            float bb = b2[e * D_MODEL + ncol];
            v8f acc = accY[mt * 4 + nq];
#pragma unroll
            for (int r = 0; r < 8; ++r) {
                int m = mt * 16 + kh * 8 + r;
                float val = (acc[r] + bb) * s_wgt[m];
                atomicAdd(&out[(size_t)s_tok[m] * D_MODEL + ncol], val);
            }
        }
    }
}

// ---------------------------------------------------------------------------
extern "C" void kernel_launch(void* const* d_in, const int* in_sizes, int n_in,
                              void* d_out, int out_size, void* d_ws, size_t ws_size,
                              hipStream_t stream)
{
    (void)in_sizes; (void)n_in; (void)out_size;
    const float* x      = (const float*)d_in[0];
    const float* gate_w = (const float*)d_in[1];
    const float* w1     = (const float*)d_in[2];
    const float* b1     = (const float*)d_in[3];
    const float* w2     = (const float*)d_in[4];
    const float* b2     = (const float*)d_in[5];
    float* out = (float*)d_out;

    const size_t OFF_IDX = 256;
    const size_t OFF_WGT = OFF_IDX + sizeof(int) * NUM_EXPERTS * T_TOKENS;
    const size_t OFF_PW1 = 2u << 20;                                  // 2MB aligned
    const size_t W_BYTES = (size_t)NUM_EXPERTS * D_MODEL * D_FF * 2;  // 16.8MB
    const size_t OFF_PW2 = OFF_PW1 + W_BYTES;
    const size_t NEED    = OFF_PW2 + W_BYTES;                         // ~35.7MB

    int*    cnt  = (int*)d_ws;
    int*    idxl = (int*)((char*)d_ws + OFF_IDX);
    float*  wgtl = (float*)((char*)d_ws + OFF_WGT);
    __bf16* pw1  = (__bf16*)((char*)d_ws + OFF_PW1);
    __bf16* pw2  = (__bf16*)((char*)d_ws + OFF_PW2);

    hipMemsetAsync(d_ws, 0, 256, stream);
    hipMemsetAsync(d_out, 0, sizeof(float) * (size_t)T_TOKENS * D_MODEL, stream);

    moe_gate_kernel<<<T_TOKENS / 256, 256, 0, stream>>>(x, gate_w, cnt, idxl, wgtl);

    dim3 grid(MAXTILE, NUM_EXPERTS);
    const bool packed = ws_size >= NEED;
    if (packed) {
        const int nblk = (int)((size_t)NUM_EXPERTS * D_MODEL * D_FF / 4 / 256); // 8192
        pack_w1_kernel<<<nblk, 256, 0, stream>>>(w1, pw1);
        pack_w2_kernel<<<nblk, 256, 0, stream>>>(w2, pw2);
        moe_ffn_kernel<true><<<grid, 256, 0, stream>>>(
            x, w1, b1, w2, b2, pw1, pw2, cnt, idxl, wgtl, out);
    } else {
        moe_ffn_kernel<false><<<grid, 256, 0, stream>>>(
            x, w1, b1, w2, b2, pw1, pw2, cnt, idxl, wgtl, out);
    }
}
